// Attention_86371792322835
// MI455X (gfx1250) — matrix operand order repaired and checked by reading, not compiled, not run on hardware
//
#include <hip/hip_runtime.h>

typedef unsigned short u16;
typedef __attribute__((ext_vector_type(16))) __bf16 v16bf;
typedef __attribute__((ext_vector_type(8)))  float  v8f;
typedef __attribute__((ext_vector_type(4)))  unsigned int u32x4;
typedef __attribute__((ext_vector_type(2)))  unsigned int u32x2;
typedef __attribute__((ext_vector_type(4)))  float f32x4;

__device__ __forceinline__ u16 f2bf(float f) {
    unsigned int u = __float_as_uint(f);
    u += 0x7FFFu + ((u >> 16) & 1u);   // round-to-nearest-even
    return (u16)(u >> 16);
}

struct Frag32 { u32x4 lo, hi; };

__device__ __forceinline__ v16bf ld_frag(const u16* p0, const u16* p1) {
    Frag32 f;
    f.lo = *(const u32x4*)p0;
    f.hi = *(const u32x4*)p1;
    return __builtin_bit_cast(v16bf, f);
}

__device__ __forceinline__ v8f wmma_bf16(v16bf a, v16bf b, v8f c) {
    return __builtin_amdgcn_wmma_f32_16x16x32_bf16(
        false, a, false, b, (short)0, c, false, false);
}

// CDNA5 async copy: global -> LDS, 16B per lane, tracked by ASYNCcnt.
__device__ __forceinline__ void async_copy_b128(void* lds_dst, const void* gsrc) {
    unsigned loff = (unsigned)(uintptr_t)lds_dst;          // wave-relative LDS offset
    unsigned long long ga = (unsigned long long)(uintptr_t)gsrc;
    asm volatile("global_load_async_to_lds_b128 %0, %1, off"
                 :: "v"(loff), "v"(ga) : "memory");
}
__device__ __forceinline__ void wait_async() {
    asm volatile("s_wait_asynccnt 0x0" ::: "memory");
}

// ---------------------------------------------------------------------------
// fp32 -> bf16 conversion (for x / activations)
// ---------------------------------------------------------------------------
__global__ __launch_bounds__(256) void cvt_f32_bf16(const float* __restrict__ in,
                                                    u16* __restrict__ out, int n4) {
    int i = blockIdx.x * 256 + threadIdx.x;
    if (i < n4) {
        f32x4 f = ((const f32x4*)in)[i];
        u32x2 o;
        o.x = (unsigned)f2bf(f.x) | ((unsigned)f2bf(f.y) << 16);
        o.y = (unsigned)f2bf(f.z) | ((unsigned)f2bf(f.w) << 16);
        ((u32x2*)out)[i] = o;
    }
}

// ---------------------------------------------------------------------------
// Fused fp32 -> bf16 + transpose for weights: Wt[n][k] = (bf16)W[k][n]
// 2048x2048, 32x32 LDS tiles, coalesced on both sides.
// ---------------------------------------------------------------------------
__global__ __launch_bounds__(256) void cvt_transpose_w(const float* __restrict__ in,
                                                       u16* __restrict__ out) {
    __shared__ u16 t[32][33];
    int k0 = blockIdx.x * 32;
    int n0 = blockIdx.y * 32;
    int lx = threadIdx.x & 31;
    int ly = threadIdx.x >> 5;
    #pragma unroll
    for (int i = 0; i < 32; i += 8)
        t[ly + i][lx] = f2bf(in[(size_t)(k0 + ly + i) * 2048 + n0 + lx]); // t[k][n]
    __syncthreads();
    #pragma unroll
    for (int i = 0; i < 32; i += 8)
        out[(size_t)(n0 + ly + i) * 2048 + k0 + lx] = t[lx][ly + i];      // out[n][k]
}

// ---------------------------------------------------------------------------
// bf16 GEMM: C[M,N] = A[M,K] * Bt[N,K]^T, fp32 accumulation.
// A row-major [M][K]; B pre-transposed (N-major [N][K]) so BOTH tiles stage
// with pure async-to-LDS b128 copies (no scatter, no VGPR round-trip).
// Block tile 128x256x32, 8 waves (2x4), wave tile 64x64 = 16 WMMA / k-step.
// Double-buffered LDS, one barrier per k-step.
// ---------------------------------------------------------------------------
__global__ __launch_bounds__(256) void gemm_bf16(const u16* __restrict__ A,
                                                 const u16* __restrict__ Bt,
                                                 float* __restrict__ Cf,
                                                 u16* __restrict__ Cb,
                                                 int M, int N, int K) {
    __shared__ u16 As[2][128][40];   // [m][k], pitch 80B = 5x16B
    __shared__ u16 Bs[2][256][40];   // [n][k]

    const int tid   = threadIdx.x;
    const int w     = tid >> 5;
    const int lane  = tid & 31;
    const int lmod  = lane & 15;
    const int lhalf = lane >> 4;
    const int wm    = w >> 2;     // 0..1
    const int wn    = w & 3;      // 0..3
    const int m0    = blockIdx.y * 128;
    const int n0    = blockIdx.x * 256;

    const v8f vzero = {0.f,0.f,0.f,0.f,0.f,0.f,0.f,0.f};
    v8f acc[4][4];
    for (int i = 0; i < 4; i++)
        for (int j = 0; j < 4; j++) acc[i][j] = vzero;

    auto stageA = [&](int buf, int k0) {        // 128x32 = 512 b128 chunks
        int c = tid * 2;
        #pragma unroll
        for (int i = 0; i < 2; i++, c++) {
            int r  = c >> 2;
            int kc = (c & 3) * 8;
            async_copy_b128(&As[buf][r][kc],
                            A + (size_t)(m0 + r) * K + k0 + kc);
        }
    };
    auto stageB = [&](int buf, int k0) {        // 256x32 = 1024 b128 chunks
        int c = tid * 4;
        #pragma unroll
        for (int i = 0; i < 4; i++, c++) {
            int r  = c >> 2;                    // = tid (4 chunks of one row)
            int kc = (c & 3) * 8;
            async_copy_b128(&Bs[buf][r][kc],
                            Bt + (size_t)(n0 + r) * K + k0 + kc);
        }
    };

    stageA(0, 0);
    stageB(0, 0);
    wait_async();
    __syncthreads();

    const int KT = K / 32;
    for (int kt = 0; kt < KT; kt++) {
        const int cur = kt & 1, nxt = cur ^ 1;
        if (kt + 1 < KT) {                      // issue next tile first
            stageA(nxt, (kt + 1) * 32);
            stageB(nxt, (kt + 1) * 32);
        }

        // Compute on current buffer: 8 frag loads (16 ds_load_b128), 16 WMMAs
        v16bf af[4], bfr[4];
        #pragma unroll
        for (int mt = 0; mt < 4; mt++) {
            int r  = wm * 64 + mt * 16 + lmod;
            int kc = lhalf * 8;
            af[mt] = ld_frag(&As[cur][r][kc], &As[cur][r][kc + 16]);
        }
        #pragma unroll
        for (int nt = 0; nt < 4; nt++) {
            int nn = wn * 64 + nt * 16 + lmod;
            int kc = lhalf * 8;
            bfr[nt] = ld_frag(&Bs[cur][nn][kc], &Bs[cur][nn][kc + 16]);
        }
        #pragma unroll
        for (int mt = 0; mt < 4; mt++)
            #pragma unroll
            for (int nt = 0; nt < 4; nt++)
                acc[mt][nt] = wmma_bf16(af[mt], bfr[nt], acc[mt][nt]);

        wait_async();
        __syncthreads();
    }

    // Epilogue: C layout — lane holds (row = v + 8*lhalf, col = lmod) per tile
    #pragma unroll
    for (int mt = 0; mt < 4; mt++) {
        #pragma unroll
        for (int nt = 0; nt < 4; nt++) {
            int col = n0 + wn * 64 + nt * 16 + lmod;
            #pragma unroll
            for (int v = 0; v < 8; v++) {
                int row = m0 + wm * 64 + mt * 16 + v + 8 * lhalf;
                float val = acc[mt][nt][v];
                if (Cf) Cf[(size_t)row * N + col] = val;
                else    Cb[(size_t)row * N + col] = f2bf(val);
            }
        }
    }
}

// ---------------------------------------------------------------------------
// Per-head transpose of V: Vt[(b*16+h)*128 + d][s] = V[b*2048+s][h*128+d]
// ---------------------------------------------------------------------------
__global__ __launch_bounds__(256) void transpose_v(const u16* __restrict__ V,
                                                   u16* __restrict__ Vt) {
    __shared__ u16 t[32][33];
    int bh = blockIdx.z;
    int b  = bh >> 4, h = bh & 15;
    int s0 = blockIdx.x * 32;
    int d0 = blockIdx.y * 32;
    int lx = threadIdx.x & 31;
    int ly = threadIdx.x >> 5;
    #pragma unroll
    for (int i = 0; i < 32; i += 8)
        t[ly + i][lx] = V[(size_t)(b * 2048 + s0 + ly + i) * 2048 + h * 128 + d0 + lx];
    __syncthreads();
    #pragma unroll
    for (int i = 0; i < 32; i += 8)
        Vt[(size_t)(bh * 128 + d0 + ly + i) * 2048 + s0 + lx] = t[lx][ly + i];
}

// ---------------------------------------------------------------------------
// Flash attention: block = (batch, head, 64-query tile), 128 threads (4 waves).
// KV tiles of 64 double-buffered in LDS via async-to-LDS copies.
// ---------------------------------------------------------------------------
__global__ __launch_bounds__(128) void flash_attn(const u16* __restrict__ Q,
                                                  const u16* __restrict__ Kg,
                                                  const u16* __restrict__ Vt,
                                                  u16* __restrict__ Ctx) {
    __shared__ u16 Ks[2][64][136];  // [key][dk], pitch 272B = 17x16B
    __shared__ u16 Vs[2][128][72];  // [dk][key], pitch 144B = 9x16B
    __shared__ u16 Ps[64][72];      // [qrow][key]

    const int bid   = blockIdx.x;
    const int qb    = bid & 31;
    const int h     = (bid >> 5) & 15;
    const int b     = bid >> 9;
    const int tid   = threadIdx.x;
    const int w     = tid >> 5;
    const int lane  = tid & 31;
    const int lmod  = lane & 15;
    const int lhalf = lane >> 4;
    const float scale = 0.08838834764831845f;  // 1/sqrt(128)

    // Q fragments (A-operand layout), resident for the whole kernel
    v16bf qf[4];
    {
        const u16* qrow =
            Q + (size_t)(b * 2048 + qb * 64 + w * 16 + lmod) * 2048 + h * 128;
        #pragma unroll
        for (int ks = 0; ks < 4; ks++) {
            int k0 = ks * 32 + lhalf * 8;
            qf[ks] = ld_frag(qrow + k0, qrow + k0 + 16);
        }
    }

    auto stageKV = [&](int buf, int kv0) {
        for (int c = tid; c < 1024; c += 128) {   // K tile: 64 x 128 bf16
            int r = c >> 4, cc = (c & 15) * 8;
            async_copy_b128(&Ks[buf][r][cc],
                Kg + (size_t)(b * 2048 + kv0 + r) * 2048 + h * 128 + cc);
        }
        for (int c = tid; c < 1024; c += 128) {   // V^T tile: 128 x 64 bf16
            int r = c >> 3, cc = (c & 7) * 8;
            async_copy_b128(&Vs[buf][r][cc],
                Vt + (size_t)((b * 16 + h) * 128 + r) * 2048 + kv0 + cc);
        }
    };

    const v8f vzero = {0.f,0.f,0.f,0.f,0.f,0.f,0.f,0.f};
    v8f o[8];
    for (int i = 0; i < 8; i++) o[i] = vzero;
    float mrow[8], lrow[8];
    for (int v = 0; v < 8; v++) { mrow[v] = -3.0e38f; lrow[v] = 0.0f; }

    stageKV(0, 0);
    wait_async();
    __syncthreads();

    for (int t = 0; t < 32; t++) {
        const int cur = t & 1, nxt = cur ^ 1;
        if (t + 1 < 32) stageKV(nxt, (t + 1) * 64);   // prefetch next tile

        // Scores: S = Q * K^T (B operand = Ks rows, naturally N-major)
        v8f sc[4];
        for (int nt = 0; nt < 4; nt++) sc[nt] = vzero;
        #pragma unroll
        for (int ks = 0; ks < 4; ks++) {
            #pragma unroll
            for (int nt = 0; nt < 4; nt++) {
                int key  = nt * 16 + lmod;
                int kofs = ks * 32 + lhalf * 8;
                v16bf kf = ld_frag(&Ks[cur][key][kofs], &Ks[cur][key][kofs + 16]);
                sc[nt] = wmma_bf16(qf[ks], kf, sc[nt]);
            }
        }

        // Online softmax per C-layout row; 16-lane half-wave reductions
        #pragma unroll
        for (int v = 0; v < 8; v++) {
            float s0 = sc[0][v] * scale, s1 = sc[1][v] * scale;
            float s2 = sc[2][v] * scale, s3 = sc[3][v] * scale;
            float mloc = fmaxf(fmaxf(s0, s1), fmaxf(s2, s3));
            #pragma unroll
            for (int mk = 1; mk < 16; mk <<= 1)
                mloc = fmaxf(mloc, __shfl_xor(mloc, mk, 32));
            float mnew  = fmaxf(mrow[v], mloc);
            float alpha = __expf(mrow[v] - mnew);
            float p0 = __expf(s0 - mnew), p1 = __expf(s1 - mnew);
            float p2 = __expf(s2 - mnew), p3 = __expf(s3 - mnew);
            sc[0][v] = p0; sc[1][v] = p1; sc[2][v] = p2; sc[3][v] = p3;
            float psum = p0 + p1 + p2 + p3;
            #pragma unroll
            for (int mk = 1; mk < 16; mk <<= 1)
                psum += __shfl_xor(psum, mk, 32);
            lrow[v] = lrow[v] * alpha + psum;
            mrow[v] = mnew;
            #pragma unroll
            for (int d = 0; d < 8; d++) o[d][v] *= alpha;
        }

        // P (C-layout) -> LDS bf16 (wave-private rows; DS stays in-order)
        #pragma unroll
        for (int nt = 0; nt < 4; nt++)
            #pragma unroll
            for (int v = 0; v < 8; v++)
                Ps[w * 16 + v + 8 * lhalf][nt * 16 + lmod] = f2bf(sc[nt][v]);

        // O += P * V
        #pragma unroll
        for (int ks = 0; ks < 2; ks++) {
            int pr = w * 16 + lmod;
            int k0 = ks * 32 + lhalf * 8;
            v16bf pf = ld_frag(&Ps[pr][k0], &Ps[pr][k0 + 16]);
            #pragma unroll
            for (int nt = 0; nt < 8; nt++) {
                int d    = nt * 16 + lmod;
                int kofs = ks * 32 + lhalf * 8;
                v16bf vf = ld_frag(&Vs[cur][d][kofs], &Vs[cur][d][kofs + 16]);
                o[nt] = wmma_bf16(pf, vf, o[nt]);
            }
        }

        wait_async();
        __syncthreads();
    }

    // Normalize and store context (bf16)
    #pragma unroll
    for (int v = 0; v < 8; v++) {
        float inv = 1.0f / lrow[v];
        int row = b * 2048 + qb * 64 + w * 16 + v + 8 * lhalf;
        u16* crow = Ctx + (size_t)row * 2048 + h * 128;
        #pragma unroll
        for (int nt = 0; nt < 8; nt++)
            crow[nt * 16 + lmod] = f2bf(o[nt][v] * inv);
    }
}

// ---------------------------------------------------------------------------
// Host-side launch
// ---------------------------------------------------------------------------
extern "C" void kernel_launch(void* const* d_in, const int* in_sizes, int n_in,
                              void* d_out, int out_size, void* d_ws, size_t ws_size,
                              hipStream_t stream) {
    (void)in_sizes; (void)n_in; (void)out_size; (void)ws_size;

    const float* x  = (const float*)d_in[0];
    const float* wq = (const float*)d_in[1];
    const float* wk = (const float*)d_in[2];
    const float* wv = (const float*)d_in[3];
    const float* wo = (const float*)d_in[4];

    const size_t XN = (size_t)8192 * 2048;
    const size_t WN = (size_t)2048 * 2048;

    u16* W   = (u16*)d_ws;
    u16* xb  = W;             // bf16 x                [8192][2048]
    u16* wqt = xb  + XN;      // bf16 Wq^T             [2048 n][2048 k]
    u16* wkt = wqt + WN;
    u16* wvt = wkt + WN;
    u16* wot = wvt + WN;
    u16* Qb  = wot + WN;      // bf16 Q                [8192][2048]
    u16* Kb  = Qb  + XN;
    u16* Vb  = Kb  + XN;
    u16* Vtb = Vb  + XN;      // bf16 V^T per head     [64*128][2048]
    u16* ctx = Vtb + XN;      // bf16 attention output [8192][2048]

    // 1) fp32 -> bf16 (x), fused convert+transpose (weights)
    {
        int n4 = (int)(XN / 4);
        cvt_f32_bf16<<<(n4 + 255) / 256, 256, 0, stream>>>(x, xb, n4);
        dim3 gw(64, 64);
        cvt_transpose_w<<<gw, 256, 0, stream>>>(wq, wqt);
        cvt_transpose_w<<<gw, 256, 0, stream>>>(wk, wkt);
        cvt_transpose_w<<<gw, 256, 0, stream>>>(wv, wvt);
        cvt_transpose_w<<<gw, 256, 0, stream>>>(wo, wot);
    }

    // 2) Q/K/V projections (bf16 out). Block tile 128x256.
    dim3 gg(2048 / 256, 8192 / 128);
    gemm_bf16<<<gg, 256, 0, stream>>>(xb, wqt, nullptr, Qb, 8192, 2048, 2048);
    gemm_bf16<<<gg, 256, 0, stream>>>(xb, wkt, nullptr, Kb, 8192, 2048, 2048);
    gemm_bf16<<<gg, 256, 0, stream>>>(xb, wvt, nullptr, Vb, 8192, 2048, 2048);

    // 3) Per-head V transpose (N-major PV operand)
    dim3 gt(2048 / 32, 128 / 32, 64);
    transpose_v<<<gt, 256, 0, stream>>>(Vb, Vtb);

    // 4) Flash attention
    flash_attn<<<2048, 128, 0, stream>>>(Qb, Kb, Vtb, ctx);

    // 5) Output projection (fp32 out)
    gemm_bf16<<<gg, 256, 0, stream>>>(ctx, wot, (float*)d_out, nullptr,
                                      8192, 2048, 2048);
}
